// GCNLayer_18330920419893
// MI455X (gfx1250) — compile-verified, matrix operand order checked by reading, not run on hardware
//
#include <hip/hip_runtime.h>
#include <hip/hip_bf16.h>

typedef __attribute__((ext_vector_type(16))) _Float16 v16h;
typedef __attribute__((ext_vector_type(8)))  float    v8f;

#define D_DIM 64

// ---------------------------------------------------------------------------
// K0: deg[i] = 1.0f  (the self edge)
// ---------------------------------------------------------------------------
__global__ void gcn_init_deg(float* __restrict__ deg, int n) {
    int i = blockIdx.x * blockDim.x + threadIdx.x;
    if (i < n) deg[i] = 1.0f;
}

// ---------------------------------------------------------------------------
// K1: deg[senders[e]] += 1  (degree over senders, per reference)
// ---------------------------------------------------------------------------
__global__ void gcn_accum_deg(const long long* __restrict__ senders,
                              float* __restrict__ deg, int nEdges) {
    int e = blockIdx.x * blockDim.x + threadIdx.x;
    if (e < nEdges) {
        atomicAdd(&deg[(int)senders[e]], 1.0f);
    }
}

// ---------------------------------------------------------------------------
// K2: h = relu(nodes @ W + b) * (1/deg)  via v_wmma_f32_16x16x32_f16.
//     Also initializes out = h (self-edge contribution).
//     256 threads = 8 waves; each wave owns one 16x16 output tile.
//     Wave w: rowTile = 2*blockIdx.x + (w>>2), colTile = w&3 (D_OUT=64 -> 4).
// ---------------------------------------------------------------------------
__global__ void __launch_bounds__(256)
gcn_gemm_relu_scale(const float* __restrict__ nodes,
                    const float* __restrict__ W,
                    const float* __restrict__ bias,
                    const float* __restrict__ deg,
                    float* __restrict__ h,
                    float* __restrict__ out,
                    int nNodes) {
    const int lane  = threadIdx.x & 31;
    const int wave  = threadIdx.x >> 5;           // 0..7
    const int rowTile = blockIdx.x * 2 + (wave >> 2);
    const int colTile = wave & 3;                 // 0..3 (16 cols each)

    if (rowTile * 16 >= nNodes) return;           // wave-uniform: EXEC stays all-1s

    const int l15   = lane & 15;
    const int half  = lane >> 4;                  // 0 or 1
    const int kbase = half * 8;
    const int aRow  = rowTile * 16 + l15;         // A-matrix row for this lane
    const int bCol  = colTile * 16 + l15;         // B-matrix col for this lane

    v8f acc = {};
    #pragma unroll
    for (int kk = 0; kk < D_DIM; kk += 32) {
        // A fragment: 16x32 f16. halves [0..7] = K kbase..kbase+7,
        //             halves [8..15] = K kbase+16..kbase+23
        v16h afrag, bfrag;
        const float* arow = nodes + (size_t)aRow * D_DIM + kk + kbase;
        #pragma unroll
        for (int j = 0; j < 8; ++j) {
            afrag[j]     = (_Float16)arow[j];
            afrag[8 + j] = (_Float16)arow[16 + j];
        }
        // B fragment: 32x16 f16, column-striped across lanes (coalesced per row)
        const float* bcol = W + (size_t)(kk + kbase) * D_DIM + bCol;
        #pragma unroll
        for (int j = 0; j < 8; ++j) {
            bfrag[j]     = (_Float16)bcol[(size_t)j * D_DIM];
            bfrag[8 + j] = (_Float16)bcol[(size_t)(16 + j) * D_DIM];
        }
        // D = A*B + C  (fp32 accumulate)
        acc = __builtin_amdgcn_wmma_f32_16x16x32_f16(
                  false, afrag, false, bfrag, (short)0, acc, false, false);
    }

    // Epilogue: +bias, ReLU, * 1/deg (deg >= 1 always), store h and out.
    const float bv = bias[bCol];
    #pragma unroll
    for (int r = 0; r < 8; ++r) {
        const int m = rowTile * 16 + r + 8 * half; // D-matrix row
        float v = acc[r] + bv;
        v = v > 0.0f ? v : 0.0f;
        v *= __builtin_amdgcn_rcpf(deg[m]);        // deg in {1,2,...}: rcp is exact enough
        h  [(size_t)m * D_DIM + bCol] = v;
        out[(size_t)m * D_DIM + bCol] = v;         // self-edge contribution
    }
}

// ---------------------------------------------------------------------------
// K3: edge scatter. 16 lanes per edge, one float4 (16B) per lane:
//     coalesced global_load_b128 gather + 4x global_atomic_add_f32 scatter.
// ---------------------------------------------------------------------------
__global__ void __launch_bounds__(256)
gcn_edge_scatter(const float4* __restrict__ h4,          // [nNodes * 16]
                 const long long* __restrict__ senders,
                 const long long* __restrict__ receivers,
                 float* __restrict__ out,
                 int nEdges) {
    const unsigned gid = blockIdx.x * blockDim.x + threadIdx.x;
    const int e = (int)(gid >> 4);
    const int g = (int)(gid & 15);                 // which 4-column chunk
    if (e >= nEdges) return;

    const int s = (int)senders[e];
    const int r = (int)receivers[e];

    const float4 v = h4[(size_t)s * 16 + g];       // 128-bit gather, L2-resident
    float* dst = out + (size_t)r * D_DIM + g * 4;
    atomicAdd(dst + 0, v.x);
    atomicAdd(dst + 1, v.y);
    atomicAdd(dst + 2, v.z);
    atomicAdd(dst + 3, v.w);
}

// ---------------------------------------------------------------------------
extern "C" void kernel_launch(void* const* d_in, const int* in_sizes, int n_in,
                              void* d_out, int out_size, void* d_ws, size_t ws_size,
                              hipStream_t stream) {
    const float*     nodes     = (const float*)d_in[0];
    const long long* senders   = (const long long*)d_in[1];
    const long long* receivers = (const long long*)d_in[2];
    const float*     W         = (const float*)d_in[3];
    const float*     bias      = (const float*)d_in[4];
    float*           out       = (float*)d_out;

    const int nNodes = in_sizes[0] / D_DIM;        // 100000
    const int nEdges = in_sizes[1];                // 1250000

    // Workspace layout: [deg: nNodes floats][h: nNodes*64 floats]
    float* deg = (float*)d_ws;
    float* h   = deg + nNodes;                     // 400000 B offset -> 16B aligned

    // K0: deg = 1
    gcn_init_deg<<<(nNodes + 255) / 256, 256, 0, stream>>>(deg, nNodes);

    // K1: deg[senders] += 1
    gcn_accum_deg<<<(nEdges + 255) / 256, 256, 0, stream>>>(senders, deg, nEdges);

    // K2: WMMA GEMM + bias + ReLU + 1/deg; out = h (self edges).
    // 8 waves/block -> 2 row-tiles x 4 col-tiles per block.
    const int rowTiles = (nNodes + 15) / 16;       // 6250
    const int blocksG  = (rowTiles + 1) / 2;       // 3125
    gcn_gemm_relu_scale<<<blocksG, 256, 0, stream>>>(nodes, W, bias, deg, h, out, nNodes);

    // K3: scatter-add over edges (16 lanes/edge).
    const long long work = (long long)nEdges * 16;
    const int blocksS = (int)((work + 255) / 256); // 78125
    gcn_edge_scatter<<<blocksS, 256, 0, stream>>>((const float4*)h, senders, receivers,
                                                  out, nEdges);
}
